// TransformerBlock_80436147520128
// MI455X (gfx1250) — matrix operand order repaired and checked by reading, not compiled, not run on hardware
//
#include <hip/hip_runtime.h>
#include <math.h>

// ---------------------------------------------------------------------------
// Types / constants
// ---------------------------------------------------------------------------
typedef _Float16 f16_t;
typedef __attribute__((ext_vector_type(16))) _Float16 v16h;
typedef __attribute__((ext_vector_type(8)))  float    v8f;

#define DEV __device__ __forceinline__

constexpr int Bc   = 2;
constexpr int Nc   = 2048;
constexpr int Ec   = 2048;
constexpr int Hc   = 16;
constexpr int Gc   = 4;
constexpr int Dc   = 128;
constexpr int HIDc = 8192;
constexpr int HD   = Hc * Dc;   // 2048
constexpr int GD   = Gc * Dc;   // 512

#define WMMA_F16(a, b, c) \
  __builtin_amdgcn_wmma_f32_16x16x32_f16(false, (a), false, (b), (short)0, (c), false, false)

// ---------------------------------------------------------------------------
// Async global->LDS copy of 16 bytes (GLOBAL_LOAD_ASYNC_TO_LDS_B128, ASYNCcnt)
// Builtin signature (from hipcc diagnostic): param1 = AS1 ptr to int4
// (GCC vector_size(16)), param2 = AS3 ptr, then imm offset, imm cpol.
// ---------------------------------------------------------------------------
typedef int v4i_raw __attribute__((vector_size(16)));
typedef __attribute__((address_space(1))) v4i_raw* as1_v4i_ptr;
typedef __attribute__((address_space(3))) v4i_raw* as3_v4i_ptr;

#if defined(__has_builtin) && __has_builtin(__builtin_amdgcn_global_load_async_to_lds_b128)
DEV void async_cp16(const void* g, void* l) {
  __builtin_amdgcn_global_load_async_to_lds_b128(
      (as1_v4i_ptr)(uintptr_t)g,
      (as3_v4i_ptr)(unsigned)(uintptr_t)l,   // LDS byte offset = addr[31:0]
      0, 0);
}
#else
DEV void async_cp16(const void* g, void* l) {
  unsigned lo = (unsigned)(uintptr_t)l;
  unsigned long long ga = (unsigned long long)(uintptr_t)g;
  asm volatile("global_load_async_to_lds_b128 %0, %1, off"
               :: "v"(lo), "v"(ga) : "memory");
}
#endif

DEV void wait_async_6() { asm volatile("s_wait_asynccnt 0x6" ::: "memory"); }
DEV void wait_async_0() { asm volatile("s_wait_asynccnt 0x0" ::: "memory"); }

// ---------------------------------------------------------------------------
// Fragment builders (CDNA5 wave32 WMMA layouts, ISA 7.12.2)
// ---------------------------------------------------------------------------
// A 16x32 f16 fragment from an LDS f32 panel (row stride 32 floats):
// lane row = lane&15, khalf = lane>>4; elems j = K khalf*16 + j.
DEV v16h lds_a_frag(const float* As, int row0, int lane) {
  const int r  = row0 + (lane & 15);
  const int kh = lane >> 4;
  const float4* p0 = reinterpret_cast<const float4*>(As + r * 32 + kh * 16);
  const float4* p1 = reinterpret_cast<const float4*>(As + r * 32 + kh * 16 + 8);
  float4 a0 = p0[0], a1 = p0[1], b0 = p1[0], b1 = p1[1];
  v16h v;
  v[0]  = (_Float16)a0.x; v[1]  = (_Float16)a0.y; v[2]  = (_Float16)a0.z; v[3]  = (_Float16)a0.w;
  v[4]  = (_Float16)a1.x; v[5]  = (_Float16)a1.y; v[6]  = (_Float16)a1.z; v[7]  = (_Float16)a1.w;
  v[8]  = (_Float16)b0.x; v[9]  = (_Float16)b0.y; v[10] = (_Float16)b0.z; v[11] = (_Float16)b0.w;
  v[12] = (_Float16)b1.x; v[13] = (_Float16)b1.y; v[14] = (_Float16)b1.z; v[15] = (_Float16)b1.w;
  return v;
}

// B 32x16 f16 fragment from an LDS f16 panel (row stride 32 halfs, N-major):
// lane col = lane&15, elems j = K (lane>>4)*16 + j.
DEV v16h lds_b_frag(const f16_t* Bs, int n0, int lane) {
  const int n  = n0 + (lane & 15);
  const int kh = lane >> 4;
  return *reinterpret_cast<const v16h*>(Bs + n * 32 + kh * 16);
}

// Direct-from-global variants (used by attention kernels)
DEV v16h load_a_f32(const float* __restrict__ A, int lda, int m0, int k0, int lane) {
  const int m  = m0 + (lane & 15);
  const int kh = lane >> 4;
  const float4* p0 = reinterpret_cast<const float4*>(A + (size_t)m * lda + k0 + kh * 16);
  const float4* p1 = reinterpret_cast<const float4*>(A + (size_t)m * lda + k0 + kh * 16 + 8);
  float4 a0 = p0[0], a1 = p0[1], b0 = p1[0], b1 = p1[1];
  v16h v;
  v[0]  = (_Float16)a0.x; v[1]  = (_Float16)a0.y; v[2]  = (_Float16)a0.z; v[3]  = (_Float16)a0.w;
  v[4]  = (_Float16)a1.x; v[5]  = (_Float16)a1.y; v[6]  = (_Float16)a1.z; v[7]  = (_Float16)a1.w;
  v[8]  = (_Float16)b0.x; v[9]  = (_Float16)b0.y; v[10] = (_Float16)b0.z; v[11] = (_Float16)b0.w;
  v[12] = (_Float16)b1.x; v[13] = (_Float16)b1.y; v[14] = (_Float16)b1.z; v[15] = (_Float16)b1.w;
  return v;
}
DEV v16h load_b_f16(const f16_t* __restrict__ Bt, int ldb, int n0, int k0, int lane) {
  const int n  = n0 + (lane & 15);
  const int kh = lane >> 4;
  return *reinterpret_cast<const v16h*>(Bt + (size_t)n * ldb + k0 + kh * 16);
}

// ---------------------------------------------------------------------------
// Panel staging (async): A 128x32 f32 (16KB), B 128x32 f16 (8KB); 6 ops/wave.
// ---------------------------------------------------------------------------
DEV void stage_panels(const float* __restrict__ A, int lda,
                      const f16_t* __restrict__ Bt, int ldb,
                      float* As, f16_t* Bs, int m0, int n0, int k0, int tid) {
#pragma unroll
  for (int i = 0; i < 4; ++i) {           // A: 1024 x 16B chunks
    int c = tid + i * 256;
    int row = c >> 3, q = (c & 7) * 4;    // q in floats
    async_cp16(A + (size_t)(m0 + row) * lda + k0 + q, As + row * 32 + q);
  }
#pragma unroll
  for (int i = 0; i < 2; ++i) {           // B: 512 x 16B chunks
    int c = tid + i * 256;
    int row = c >> 2, q = (c & 3) * 8;    // q in halfs
    async_cp16(Bt + (size_t)(n0 + row) * ldb + k0 + q, Bs + row * 32 + q);
  }
}

// ---------------------------------------------------------------------------
// Tiled GEMM: C[M,N] = A[M,K](f32) x Bt[N,K](f16) (+ optional residual)
// block 256 thr = 8 waves (4M x 2N); block tile 128x128x32, wave tile 32x64.
// Double-buffered async LDS staging.
// ---------------------------------------------------------------------------
__global__ void __launch_bounds__(256)
wmma_gemm_kernel(const float* __restrict__ A, int lda,
                 const f16_t* __restrict__ Bt, int ldb,
                 float* __restrict__ C, int ldc,
                 const float* __restrict__ R, int K) {
  __shared__ float As[2][128 * 32];
  __shared__ f16_t Bs[2][128 * 32];
  const int tid = threadIdx.x;
  const int lane = tid & 31, wave = tid >> 5;
  const int mw = wave >> 1;                 // 0..3
  const int nw = wave & 1;                  // 0..1
  const int m0 = blockIdx.y * 128;
  const int n0 = blockIdx.x * 128;

  v8f acc[2][4] = {};
  const int KT = K >> 5;

  stage_panels(A, lda, Bt, ldb, As[0], Bs[0], m0, n0, 0, tid);
  for (int kt = 0; kt < KT; ++kt) {
    const int cur = kt & 1;
    if (kt + 1 < KT) {
      stage_panels(A, lda, Bt, ldb, As[cur ^ 1], Bs[cur ^ 1], m0, n0, (kt + 1) * 32, tid);
      wait_async_6();                       // stage kt done, stage kt+1 in flight
    } else {
      wait_async_0();
    }
    __syncthreads();
    v16h af[2], bf[4];
#pragma unroll
    for (int mi = 0; mi < 2; ++mi) af[mi] = lds_a_frag(As[cur], mw * 32 + mi * 16, lane);
#pragma unroll
    for (int ni = 0; ni < 4; ++ni) bf[ni] = lds_b_frag(Bs[cur], nw * 64 + ni * 16, lane);
#pragma unroll
    for (int mi = 0; mi < 2; ++mi)
#pragma unroll
      for (int ni = 0; ni < 4; ++ni)
        acc[mi][ni] = WMMA_F16(af[mi], bf[ni], acc[mi][ni]);
    __syncthreads();
  }

#pragma unroll
  for (int mi = 0; mi < 2; ++mi) {
    const int mb = m0 + mw * 32 + mi * 16 + (lane >> 4) * 8;
#pragma unroll
    for (int ni = 0; ni < 4; ++ni) {
      const int n = n0 + nw * 64 + ni * 16 + (lane & 15);
#pragma unroll
      for (int j = 0; j < 8; ++j) {
        const size_t off = (size_t)(mb + j) * ldc + n;
        C[off] = R ? (acc[mi][ni][j] + R[off]) : acc[mi][ni][j];
      }
    }
  }
}

// ---------------------------------------------------------------------------
// Tiled GeGLU GEMM: H[m,n] = U[m,n] * gelu_tanh(Gt[m,n]),
//   U = X1 @ w1[:, 0:HID], Gt = X1 @ w1[:, HID:2*HID]
// block tile 128(M) x 64(N) with two B panels (u and gate); wave tile 32x32x2.
// ---------------------------------------------------------------------------
__global__ void __launch_bounds__(256)
geglu_gemm_kernel(const float* __restrict__ X, const f16_t* __restrict__ W1t,
                  float* __restrict__ Hout) {
  __shared__ float As[2][128 * 32];
  __shared__ f16_t Bus[2][64 * 32];
  __shared__ f16_t Bgs[2][64 * 32];
  const int tid = threadIdx.x;
  const int lane = tid & 31, wave = tid >> 5;
  const int mw = wave >> 1;                 // 0..3
  const int nw = wave & 1;                  // 0..1
  const int m0 = blockIdx.y * 128;
  const int n0 = blockIdx.x * 64;

  v8f au[2][2] = {}, ag[2][2] = {};
  const int KT = Ec >> 5;                   // 64

  auto stage = [&](float* Asb, f16_t* Bub, f16_t* Bgb, int k0) {
#pragma unroll
    for (int i = 0; i < 4; ++i) {           // A panel
      int c = tid + i * 256;
      int row = c >> 3, q = (c & 7) * 4;
      async_cp16(X + (size_t)(m0 + row) * Ec + k0 + q, Asb + row * 32 + q);
    }
    {                                       // u panel: 64x32 halfs
      int row = tid >> 2, q = (tid & 3) * 8;
      async_cp16(W1t + (size_t)(n0 + row) * Ec + k0 + q, Bub + row * 32 + q);
    }
    {                                       // gate panel
      int row = tid >> 2, q = (tid & 3) * 8;
      async_cp16(W1t + (size_t)(HIDc + n0 + row) * Ec + k0 + q, Bgb + row * 32 + q);
    }
  };

  stage(As[0], Bus[0], Bgs[0], 0);
  for (int kt = 0; kt < KT; ++kt) {
    const int cur = kt & 1;
    if (kt + 1 < KT) {
      stage(As[cur ^ 1], Bus[cur ^ 1], Bgs[cur ^ 1], (kt + 1) * 32);
      wait_async_6();
    } else {
      wait_async_0();
    }
    __syncthreads();
    v16h af[2], bu[2], bg[2];
#pragma unroll
    for (int mi = 0; mi < 2; ++mi) af[mi] = lds_a_frag(As[cur], mw * 32 + mi * 16, lane);
#pragma unroll
    for (int ni = 0; ni < 2; ++ni) {
      bu[ni] = lds_b_frag(Bus[cur], nw * 32 + ni * 16, lane);
      bg[ni] = lds_b_frag(Bgs[cur], nw * 32 + ni * 16, lane);
    }
#pragma unroll
    for (int mi = 0; mi < 2; ++mi)
#pragma unroll
      for (int ni = 0; ni < 2; ++ni) {
        au[mi][ni] = WMMA_F16(af[mi], bu[ni], au[mi][ni]);
        ag[mi][ni] = WMMA_F16(af[mi], bg[ni], ag[mi][ni]);
      }
    __syncthreads();
  }

#pragma unroll
  for (int mi = 0; mi < 2; ++mi) {
    const int mb = m0 + mw * 32 + mi * 16 + (lane >> 4) * 8;
#pragma unroll
    for (int ni = 0; ni < 2; ++ni) {
      const int n = n0 + nw * 32 + ni * 16 + (lane & 15);
#pragma unroll
      for (int j = 0; j < 8; ++j) {
        float u = au[mi][ni][j], gt = ag[mi][ni][j];
        float inner = 0.7978845608028654f * (gt + 0.044715f * gt * gt * gt);
        float gl = 0.5f * gt * (1.f + tanhf(inner));
        Hout[(size_t)(mb + j) * HIDc + n] = u * gl;
      }
    }
  }
}

// ---------------------------------------------------------------------------
// Weight transpose + f32->f16 convert:  Wt[n*K + k] = (f16) W[k*N + n]
// ---------------------------------------------------------------------------
__global__ void transpose_cvt_kernel(const float* __restrict__ W, f16_t* __restrict__ Wt,
                                     int K, int N) {
  __shared__ float tile[32][33];
  const int kb = blockIdx.y * 32, nb = blockIdx.x * 32;
  const int tx = threadIdx.x & 31, ty = threadIdx.x >> 5;
  for (int i = ty; i < 32; i += 8) {
    int k = kb + i, n = nb + tx;
    tile[i][tx] = (k < K && n < N) ? W[(size_t)k * N + n] : 0.f;
  }
  __syncthreads();
  for (int i = ty; i < 32; i += 8) {
    int n = nb + i, k = kb + tx;
    if (n < N && k < K) Wt[(size_t)n * K + k] = (f16_t)tile[tx][i];
  }
}

// ---------------------------------------------------------------------------
// RoPE: xr = x*cos + rotate_half(x)*sin  (q_r == k_r); also k_flat/v_flat out
// ---------------------------------------------------------------------------
__global__ void rope_kernel(const float* __restrict__ x, float* __restrict__ xr,
                            float* __restrict__ kout, float* __restrict__ vout) {
  const size_t idx = (size_t)blockIdx.x * blockDim.x + threadIdx.x;
  const int e = (int)(idx % Ec);
  const int n = (int)((idx / Ec) % Nc);
  const int d = e & (Dc - 1);
  const int i = d & 63;
  const float inv = __powf(10000.f, -(float)(2 * i) * (1.0f / (float)Dc));
  const float ang = (float)n * inv;
  float s, c;
  __sincosf(ang, &s, &c);
  const float v = x[idx];
  const float other = (d < Dc / 2) ? -x[idx + Dc / 2] : x[idx - Dc / 2];
  const float r = v * c + other * s;
  xr[idx] = r;
  kout[idx] = r;
  vout[idx] = v;
}

// ---------------------------------------------------------------------------
// Pack K/V heads: kh16[b][g][n][d] = kh, vh16t[b][g][d][n] = vh (f32 -> f16)
// ---------------------------------------------------------------------------
__global__ void kv_pack_kernel(const float* __restrict__ kh, const float* __restrict__ vh,
                               f16_t* __restrict__ kh16, f16_t* __restrict__ vh16t) {
  const size_t idx = (size_t)blockIdx.x * blockDim.x + threadIdx.x;
  const int col = (int)(idx % GD);
  const int d = col & (Dc - 1);
  const int g = col >> 7;
  const int n = (int)((idx / GD) % Nc);
  const int b = (int)(idx / ((size_t)Nc * GD));
  kh16[((size_t)(b * Gc + g) * Nc + n) * Dc + d] = (f16_t)kh[idx];
  vh16t[((size_t)(b * Gc + g) * Dc + d) * Nc + n] = (f16_t)vh[idx];
}

// ---------------------------------------------------------------------------
// scores(q,k) = scale * sum_d qh(q,d)*kh(k,d); raw scores -> attn region
// ---------------------------------------------------------------------------
__global__ void scores_kernel(const float* __restrict__ qh, const f16_t* __restrict__ kh16,
                              float* __restrict__ attn) {
  const int lane = threadIdx.x & 31, wave = threadIdx.x >> 5;
  const int bh = blockIdx.z, b = bh >> 4, h = bh & 15, g = h >> 2;
  const int m0 = blockIdx.y * 16;
  const int n0 = (blockIdx.x * 8 + wave) * 16;
  const float* A = qh + (size_t)b * Nc * HD + h * Dc;
  const f16_t* Bt = kh16 + (size_t)(b * Gc + g) * Nc * Dc;
  v8f acc = {};
#pragma unroll
  for (int k0 = 0; k0 < Dc; k0 += 32)
    acc = WMMA_F16(load_a_f32(A, HD, m0, k0, lane), load_b_f16(Bt, Dc, n0, k0, lane), acc);
  float* O = attn + (size_t)bh * Nc * Nc;
  const int n = n0 + (lane & 15);
  const int mb = m0 + (lane >> 4) * 8;
  const float scale = 0.08838834764831845f;  // 1/sqrt(128)
#pragma unroll
  for (int j = 0; j < 8; ++j) O[(size_t)(mb + j) * Nc + n] = acc[j] * scale;
}

// ---------------------------------------------------------------------------
// In-place row softmax over 2048 elements, one block (256 thr) per row
// ---------------------------------------------------------------------------
__global__ void softmax_kernel(float* __restrict__ attn) {
  float* p = attn + (size_t)blockIdx.x * Nc;
  const int tid = threadIdx.x;
  __shared__ float sd[256];
  float v[8];
  float m = -3.4e38f;
#pragma unroll
  for (int i = 0; i < 8; ++i) { v[i] = p[tid + i * 256]; m = fmaxf(m, v[i]); }
  sd[tid] = m; __syncthreads();
  for (int s = 128; s > 0; s >>= 1) { if (tid < s) sd[tid] = fmaxf(sd[tid], sd[tid + s]); __syncthreads(); }
  m = sd[0]; __syncthreads();
  float sum = 0.f;
#pragma unroll
  for (int i = 0; i < 8; ++i) { v[i] = __expf(v[i] - m); sum += v[i]; }
  sd[tid] = sum; __syncthreads();
  for (int s = 128; s > 0; s >>= 1) { if (tid < s) sd[tid] += sd[tid + s]; __syncthreads(); }
  const float inv = 1.f / sd[0];
#pragma unroll
  for (int i = 0; i < 8; ++i) p[tid + i * 256] = v[i] * inv;
}

// ---------------------------------------------------------------------------
// ctx(q,d) = sum_k attn(q,k) * vh(k,d)  -> ctx[b*N+q][h*D+d]
// ---------------------------------------------------------------------------
__global__ void ctx_kernel(const float* __restrict__ attn, const f16_t* __restrict__ vh16t,
                           float* __restrict__ ctx) {
  const int lane = threadIdx.x & 31, wave = threadIdx.x >> 5;
  const int bh = blockIdx.z, b = bh >> 4, h = bh & 15, g = h >> 2;
  const int m0 = blockIdx.y * 16;
  const int n0 = wave * 16;
  const float* A = attn + (size_t)bh * Nc * Nc;
  const f16_t* Bt = vh16t + (size_t)(b * Gc + g) * Dc * Nc;
  v8f acc = {};
  for (int k0 = 0; k0 < Nc; k0 += 32)
    acc = WMMA_F16(load_a_f32(A, Nc, m0, k0, lane), load_b_f16(Bt, Nc, n0, k0, lane), acc);
  float* O = ctx + (size_t)b * Nc * HD + h * Dc;
  const int n = n0 + (lane & 15);
  const int mb = m0 + (lane >> 4) * 8;
#pragma unroll
  for (int j = 0; j < 8; ++j) O[(size_t)(mb + j) * HD + n] = acc[j];
}

// ---------------------------------------------------------------------------
// RMSNorm: o = y * rsqrt(mean(y^2) + eps) * w
// ---------------------------------------------------------------------------
__global__ void rmsnorm_kernel(const float* __restrict__ y, const float* __restrict__ w,
                               float* __restrict__ o) {
  const float* p = y + (size_t)blockIdx.x * Ec;
  float* q = o + (size_t)blockIdx.x * Ec;
  const int tid = threadIdx.x;
  __shared__ float sd[256];
  float v[8];
  float s = 0.f;
#pragma unroll
  for (int i = 0; i < 8; ++i) { v[i] = p[tid + i * 256]; s += v[i] * v[i]; }
  sd[tid] = s; __syncthreads();
  for (int st = 128; st > 0; st >>= 1) { if (tid < st) sd[tid] += sd[tid + st]; __syncthreads(); }
  const float r = rsqrtf(sd[0] * (1.0f / (float)Ec) + 1e-6f);
#pragma unroll
  for (int i = 0; i < 8; ++i) q[tid + i * 256] = v[i] * r * w[tid + i * 256];
}

// ---------------------------------------------------------------------------
// Host-side orchestration
// ---------------------------------------------------------------------------
extern "C" void kernel_launch(void* const* d_in, const int* in_sizes, int n_in,
                              void* d_out, int out_size, void* d_ws, size_t ws_size,
                              hipStream_t stream) {
  const float* x   = (const float*)d_in[0];
  const float* wq  = (const float*)d_in[1];
  const float* wk  = (const float*)d_in[2];
  const float* wv  = (const float*)d_in[3];
  const float* wo  = (const float*)d_in[4];
  const float* w1  = (const float*)d_in[5];
  const float* w2  = (const float*)d_in[6];
  const float* n1w = (const float*)d_in[7];
  const float* n2w = (const float*)d_in[8];

  const size_t M = (size_t)Bc * Nc;  // 4096 rows

  // ---- workspace layout (f32 region, then f16 region) ----
  float* f   = (float*)d_ws;
  float* XR  = f;                 // M*E   roped x (dead after kh GEMM)
  float* QH  = XR + M * Ec;       // M*E   (aliased later as Y2)
  float* KH  = QH + M * Ec;       // M*GD
  float* VH  = KH + M * GD;       // M*GD
  float* Y1  = VH + M * GD;       // M*E
  float* X1  = Y1 + M * Ec;       // M*E
  float* Hb  = X1 + M * Ec;       // M*HID
  f16_t* hp  = (f16_t*)(Hb + M * (size_t)HIDc);
  f16_t* KH16  = hp;                          // B*G*N*D halfs
  f16_t* VH16T = KH16 + M * GD;
  f16_t* WQt = VH16T + M * GD;                // E*E
  f16_t* WKt = WQt + (size_t)Ec * Ec;         // E*GD
  f16_t* WVt = WKt + (size_t)Ec * GD;
  f16_t* WOt = WVt + (size_t)Ec * GD;         // E*E
  f16_t* W1t = WOt + (size_t)Ec * Ec;         // 2*HID*E
  f16_t* W2t = W1t + (size_t)Ec * 2 * HIDc;   // HID*E
  float* CTX = XR;  // alias (XR dead)
  float* Y2  = QH;  // alias (QH dead after scores)

  // ---- output layout: x2 | k_flat | v_flat | attn ----
  float* out_x2   = (float*)d_out;
  float* out_k    = out_x2 + M * Ec;
  float* out_v    = out_k + M * Ec;
  float* out_attn = out_v + M * Ec;

  // 1) weight transpose+convert
  auto tcvt = [&](const float* W, f16_t* Wt, int K, int N) {
    dim3 g((N + 31) / 32, (K + 31) / 32);
    transpose_cvt_kernel<<<g, 256, 0, stream>>>(W, Wt, K, N);
  };
  tcvt(wq, WQt, Ec, Ec);
  tcvt(wk, WKt, Ec, GD);
  tcvt(wv, WVt, Ec, GD);
  tcvt(wo, WOt, Ec, Ec);
  tcvt(w1, W1t, Ec, 2 * HIDc);
  tcvt(w2, W2t, HIDc, Ec);

  // 2) RoPE (+ k_flat / v_flat outputs)
  rope_kernel<<<(unsigned)(M * Ec / 256), 256, 0, stream>>>(x, XR, out_k, out_v);

  // 3) projections (tiled async WMMA GEMM; block tile 128x128)
  auto gemm = [&](const float* A, int lda, const f16_t* Bt, int ldb, float* C, int ldc,
                  const float* R, int K, int Nn) {
    dim3 g(Nn / 128, (unsigned)(M / 128));
    wmma_gemm_kernel<<<g, 256, 0, stream>>>(A, lda, Bt, ldb, C, ldc, R, K);
  };
  gemm(XR, Ec, WQt, Ec, QH, Ec, nullptr, Ec, Ec);   // qh
  gemm(XR, Ec, WKt, Ec, KH, GD, nullptr, Ec, GD);   // kh
  gemm(x,  Ec, WVt, Ec, VH, GD, nullptr, Ec, GD);   // vh

  // 4) pack K/V to f16 (V transposed)
  kv_pack_kernel<<<(unsigned)(M * GD / 256), 256, 0, stream>>>(KH, VH, KH16, VH16T);

  // 5) raw scores -> attn region
  scores_kernel<<<dim3(Nc / 128, Nc / 16, Bc * Hc), 256, 0, stream>>>(QH, KH16, out_attn);

  // 6) softmax in place
  softmax_kernel<<<(unsigned)((size_t)Bc * Hc * Nc), 256, 0, stream>>>(out_attn);

  // 7) ctx = attn @ vh
  ctx_kernel<<<dim3(1, Nc / 16, Bc * Hc), 256, 0, stream>>>(out_attn, VH16T, CTX);

  // 8) y1 = ctx @ wo + x ; x1 = rmsnorm(y1)
  gemm(CTX, Ec, WOt, Ec, Y1, Ec, x, Ec, Ec);
  rmsnorm_kernel<<<(unsigned)M, 256, 0, stream>>>(Y1, n1w, X1);

  // 9) GeGLU FFN (fused u/gate tiled GEMM)
  geglu_gemm_kernel<<<dim3(HIDc / 64, (unsigned)(M / 128)), 256, 0, stream>>>(X1, W1t, Hb);

  // 10) y2 = h @ w2 + x1 ; x2 = rmsnorm(y2) -> output
  gemm(Hb, HIDc, W2t, HIDc, Y2, Ec, X1, HIDc, Ec);
  rmsnorm_kernel<<<(unsigned)M, 256, 0, stream>>>(Y2, n2w, out_x2);
}
